// SimplePPOPolicyGNN_1614907703909
// MI455X (gfx1250) — compile-verified, matrix operand order checked by reading, not run on hardware
//
#include <hip/hip_runtime.h>
#include <hip/hip_bf16.h>
#include <math.h>

typedef __attribute__((ext_vector_type(16))) _Float16 v16h;
typedef __attribute__((ext_vector_type(8)))  float    v8f;

#define NN      16384
#define EE      524288
#define FIN     128
#define HH      256
#define AA      64
#define MASK_W  (NN * (NN / 32))          // 8,388,608 u32 words = 32 MB

// ---------------- init: zero dedupe bitmask, aggX = X (self loop), deg = 1, g = 0
__global__ void init_ws(unsigned* __restrict__ mask,
                        float* __restrict__ aggX,
                        const float* __restrict__ X,
                        float* __restrict__ deg,
                        float* __restrict__ g) {
    int i = blockIdx.x * blockDim.x + threadIdx.x;       // exactly N*FIN = 2,097,152 threads
    #pragma unroll
    for (int w = i; w < MASK_W; w += NN * FIN) mask[w] = 0u;
    aggX[i] = X[i];
    if (i < NN)  deg[i] = 1.0f;
    if (i < HH)  g[i]   = 0.0f;
}

// ---------------- edge aggregation: 1 wave per edge, dedupe with bitmask (matches .set semantics)
__global__ void edge_agg(const int* __restrict__ ei,
                         const float* __restrict__ X,
                         unsigned* __restrict__ mask,
                         float* __restrict__ deg,
                         float* __restrict__ aggX) {
    int lane = threadIdx.x & 31;
    int e = (blockIdx.x * blockDim.x + threadIdx.x) >> 5;
    if (e >= EE) return;
    int s = ei[e];
    int d = ei[EE + e];
    int isNew = 0;
    if (lane == 0) {
        unsigned word = ((unsigned)s << 9) | ((unsigned)d >> 5);   // s * (N/32) + d/32
        unsigned bit  = 1u << (d & 31);
        unsigned old  = atomicOr(mask + word, bit);
        isNew = (old & bit) == 0;
        if (isNew) atomicAdd(deg + s, 1.0f);
    }
    isNew = __builtin_amdgcn_readfirstlane(isNew);
    if (!isNew) return;
    const float* xs = X    + (size_t)d * FIN;
    float*       as = aggX + (size_t)s * FIN;
    #pragma unroll
    for (int f = lane; f < FIN; f += 32) atomicAdd(as + f, xs[f]);
}

// ---------------- row-normalize and convert to f16 for the WMMA GEMM
__global__ void normalize_convert(const float* __restrict__ aggX,
                                  const float* __restrict__ deg,
                                  _Float16* __restrict__ Af16) {
    int i = blockIdx.x * blockDim.x + threadIdx.x;       // N*FIN threads
    Af16[i] = (_Float16)(aggX[i] / deg[i >> 7]);
}

__global__ void convert_w(const float* __restrict__ W, _Float16* __restrict__ Wf16) {
    int i = blockIdx.x * blockDim.x + threadIdx.x;
    if (i < HH * FIN) Wf16[i] = (_Float16)W[i];
}

// ---------------- GEMM h = relu(aggX_norm @ W_gnn^T + b) fused with mean-pool into g[256]
// One wave per 16x16 output tile; K=128 -> 4 x v_wmma_f32_16x16x32_f16.
__global__ void gemm_relu_meanpool(const _Float16* __restrict__ A,  // [N, 128]
                                   const _Float16* __restrict__ B,  // W_gnn f16 [256, 128] row-major
                                   const float* __restrict__ bias,
                                   float* __restrict__ g) {
    int wave  = (blockIdx.x * blockDim.x + threadIdx.x) >> 5;   // 0..16383
    int lane  = threadIdx.x & 31;
    int tileM = wave >> 4;                                      // 0..1023
    int tileN = wave & 15;                                      // 0..15
    int rc    = lane & 15;                                      // row (A) / col (B)
    int hiK   = (lane >> 4) << 3;                               // 0 or 8 (ISA A/B f16 layout)

    const _Float16* arow = A + (size_t)(tileM * 16 + rc) * FIN;
    const _Float16* brow = B + (size_t)(tileN * 16 + rc) * FIN; // col n of B == row n of W_gnn

    v8f acc = {};
    #pragma unroll
    for (int k0 = 0; k0 < FIN; k0 += 32) {
        v16h a, b;
        #pragma unroll
        for (int t = 0; t < 8; ++t) {
            a[t]     = arow[k0 + hiK + t];
            a[t + 8] = arow[k0 + 16 + hiK + t];
            b[t]     = brow[k0 + hiK + t];
            b[t + 8] = brow[k0 + 16 + hiK + t];
        }
        acc = __builtin_amdgcn_wmma_f32_16x16x32_f16(
            /*neg_a=*/false, a, /*neg_b=*/false, b,
            /*c_mod=*/(short)0, acc, /*reuse_a=*/false, /*reuse_b=*/false);
    }
    // D element r -> h[tileM*16 + r + hiK][tileN*16 + rc]; bias+relu then column partial-sum
    int n = tileN * 16 + rc;
    float bn = bias[n];
    float s = 0.0f;
    #pragma unroll
    for (int r = 0; r < 8; ++r) {
        float v = acc[r] + bn;
        s += v > 0.0f ? v : 0.0f;
    }
    atomicAdd(g + n, s * (1.0f / (float)NN));
}

// ---------------- policy / value heads + softmax (single block)
__global__ void head_kernel(const float* __restrict__ g,
                            const float* __restrict__ W_a1, const float* __restrict__ b_a1,
                            const float* __restrict__ W_a2, const float* __restrict__ b_a2,
                            const float* __restrict__ W_c1, const float* __restrict__ b_c1,
                            const float* __restrict__ W_c2, const float* __restrict__ b_c2,
                            float* __restrict__ out) {
    __shared__ float sg[HH];
    __shared__ float ah[HH / 2];
    __shared__ float ch[HH / 2];
    __shared__ float logits[AA];
    __shared__ float red[2];
    int t = threadIdx.x;
    sg[t] = g[t];
    __syncthreads();
    if (t < HH / 2) {
        float sa = b_a1[t], sc = b_c1[t];
        for (int k = 0; k < HH; ++k) {
            float gv = sg[k];
            sa = fmaf(gv, W_a1[t * HH + k], sa);
            sc = fmaf(gv, W_c1[t * HH + k], sc);
        }
        ah[t] = sa > 0.0f ? sa : 0.0f;
        ch[t] = sc > 0.0f ? sc : 0.0f;
    }
    __syncthreads();
    if (t < AA) {
        float s = b_a2[t];
        for (int k = 0; k < HH / 2; ++k) s = fmaf(ah[k], W_a2[t * (HH / 2) + k], s);
        logits[t] = s;
    }
    if (t == HH - 1) {
        float v = b_c2[0];
        for (int k = 0; k < HH / 2; ++k) v = fmaf(ch[k], W_c2[k], v);
        out[AA] = v;                       // value scalar
    }
    __syncthreads();
    if (t == 0) {
        float mx = logits[0];
        for (int k = 1; k < AA; ++k) mx = fmaxf(mx, logits[k]);
        float den = 0.0f;
        for (int k = 0; k < AA; ++k) den += __expf(logits[k] - mx);
        red[0] = mx; red[1] = 1.0f / den;
    }
    __syncthreads();
    if (t < AA) out[t] = __expf(logits[t] - red[0]) * red[1];
}

extern "C" void kernel_launch(void* const* d_in, const int* in_sizes, int n_in,
                              void* d_out, int out_size, void* d_ws, size_t ws_size,
                              hipStream_t stream) {
    const float* X     = (const float*)d_in[0];
    const int*   ei    = (const int*)  d_in[1];
    const float* W_gnn = (const float*)d_in[2];
    const float* b_gnn = (const float*)d_in[3];
    const float* W_a1  = (const float*)d_in[4];
    const float* b_a1  = (const float*)d_in[5];
    const float* W_a2  = (const float*)d_in[6];
    const float* b_a2  = (const float*)d_in[7];
    const float* W_c1  = (const float*)d_in[8];
    const float* b_c1  = (const float*)d_in[9];
    const float* W_c2  = (const float*)d_in[10];
    const float* b_c2  = (const float*)d_in[11];
    float* out = (float*)d_out;

    char* ws = (char*)d_ws;
    size_t off = 0;
    unsigned*  mask  = (unsigned*)(ws + off); off += (size_t)MASK_W * 4;        // 32 MB
    float*     aggX  = (float*)   (ws + off); off += (size_t)NN * FIN * 4;      // 8 MB
    float*     deg   = (float*)   (ws + off); off += (size_t)NN * 4;
    float*     g     = (float*)   (ws + off); off += (size_t)HH * 4;
    _Float16*  Af16  = (_Float16*)(ws + off); off += (size_t)NN * FIN * 2;      // 4 MB
    _Float16*  Wf16  = (_Float16*)(ws + off); off += (size_t)HH * FIN * 2;
    if (off > ws_size) return;  // workspace too small; nothing safe to do

    // 1) init: mask=0, aggX=X (self loop), deg=1, g=0
    init_ws<<<(NN * FIN) / 256, 256, 0, stream>>>(mask, aggX, X, deg, g);

    // 2) deduped edge aggregation (one wave per edge)
    edge_agg<<<(EE * 32) / 256, 256, 0, stream>>>(ei, X, mask, deg, aggX);

    // 3) row-normalize + f16 convert; convert W_gnn
    normalize_convert<<<(NN * FIN) / 256, 256, 0, stream>>>(aggX, deg, Af16);
    convert_w<<<(HH * FIN + 255) / 256, 256, 0, stream>>>(W_gnn, Wf16);

    // 4) WMMA GEMM + bias + relu + mean-pool into g  (16384 waves, 4 waves/block)
    gemm_relu_meanpool<<<(NN / 16) * (HH / 16) / 4, 128, 0, stream>>>(Af16, Wf16, b_gnn, g);

    // 5) heads + softmax -> out[0..63] probs, out[64] value
    head_kernel<<<1, HH, 0, stream>>>(g, W_a1, b_a1, W_a2, b_a2, W_c1, b_c1, W_c2, b_c2, out);
}
